// GCNN_mutual_attention_26688926777487
// MI455X (gfx1250) — matrix operand order repaired and checked by reading, not compiled, not run on hardware
//
#include <hip/hip_runtime.h>

// ---------------------------------------------------------------------------
// CDNA5 (gfx1250) implementation of the GCN + mutual-attention reference.
// Dense layers use v_wmma_f32_16x16x32_bf16 (wave32).
//   - gemm_bf16_fast: A (bf16, pre-converted), aligned M%16/N%32/K%32,
//     branch-free K loop, 2 output tiles (16x32) per wave.
//   - gemm_bf16_gen : fp32 A, guarded tails (only for W_red: K=80, N=30).
// ---------------------------------------------------------------------------

typedef __attribute__((ext_vector_type(16))) __bf16 v16bf;
typedef __attribute__((ext_vector_type(8)))  __bf16 v8bf;
typedef __attribute__((ext_vector_type(8)))  float  v8f;

#define NEG_SLOPE 0.01f
#define LN_EPS 1e-5f

static inline int ceil_div_h(long long a, long long b) { return (int)((a + b - 1) / b); }

__device__ __forceinline__ __bf16 f2bf(float f) {
  union { float f; unsigned u; } a; a.f = f;
  unsigned r = a.u + 0x7FFFu + ((a.u >> 16) & 1u);   // round-to-nearest-even
  union { unsigned short s; __bf16 b; } c; c.s = (unsigned short)(r >> 16);
  return c.b;
}
__device__ __forceinline__ unsigned short f2us(float f) {
  union { float f; unsigned u; } a; a.f = f;
  unsigned r = a.u + 0x7FFFu + ((a.u >> 16) & 1u);
  return (unsigned short)(r >> 16);
}
__device__ __forceinline__ __bf16 us2bf(unsigned short u) {
  union { unsigned short s; __bf16 b; } c; c.s = u; return c.b;
}

__device__ __forceinline__ void store_tile(float* __restrict__ C,
                                           const float* __restrict__ bias,
                                           const float* __restrict__ residual,
                                           v8f acc, int mt, int cb, int lane,
                                           int ldc, int act)
{
  float bvv = bias ? bias[cb] : 0.f;
#pragma unroll
  for (int v = 0; v < 8; ++v) {
    int cr = mt * 16 + v + ((lane >> 4) << 3);
    float val = acc[v] + bvv;
    if (act == 1)      val = (val >= 0.f) ? val : NEG_SLOPE * val;
    else if (act == 2) val = (val > 0.f) ? val : 0.f;
    if (residual) val += residual[(size_t)cr * ldc + cb];
    C[(size_t)cr * ldc + cb] = val;
  }
}

// ---------------------------------------------------------------------------
// Fast GEMM: C[M,N] = act(Ab[M,K] @ Bt[N,K]^T + bias) + residual
// Preconditions: M%16==0, N%32==0, K%32==0. Ab/Bt are bf16.
// One wave computes a 16x32 output strip (two 16x16 WMMA tiles sharing A).
// ---------------------------------------------------------------------------
__global__ void __launch_bounds__(256)
gemm_bf16_fast(const unsigned short* __restrict__ Ain,
               const unsigned short* __restrict__ Btin,
               const float* __restrict__ bias, const float* __restrict__ residual,
               float* __restrict__ C, int M, int N, int K, int ldc, int act)
{
  int wave = (int)((blockIdx.x * blockDim.x + threadIdx.x) >> 5);
  int lane = threadIdx.x & 31;
  int ng2  = N >> 5;                        // groups of two 16-col tiles
  if (wave >= (M >> 4) * ng2) return;       // whole-wave exit; EXEC stays all-1s
  int mt = wave / ng2, ng = wave % ng2;

  const __bf16* A  = (const __bf16*)Ain;
  const __bf16* Bt = (const __bf16*)Btin;

  int ra    = mt * 16 + (lane & 15);
  int abase = (lane >> 4) * 16;             // A K sub-base (0 or 16)
  int cb0   = ng * 32 + (lane & 15);
  int cb1   = cb0 + 16;
  int bbase = (lane >> 4) * 16;             // B K sub-base (0 or 16)

  const __bf16* ap  = A  + (size_t)ra  * K + abase;
  const __bf16* bp0 = Bt + (size_t)cb0 * K + bbase;
  const __bf16* bp1 = Bt + (size_t)cb1 * K + bbase;

  v8f acc0 = {0.f,0.f,0.f,0.f,0.f,0.f,0.f,0.f};
  v8f acc1 = {0.f,0.f,0.f,0.f,0.f,0.f,0.f,0.f};

#pragma unroll 2
  for (int kt = 0; kt < K; kt += 32) {
    v8bf a0 = *(const v8bf*)(ap + kt);        // K = abase+0..7
    v8bf a1 = *(const v8bf*)(ap + kt + 8);    // K = abase+8..15
    v16bf av = __builtin_shufflevector(a0, a1, 0,1,2,3,4,5,6,7,
                                               8,9,10,11,12,13,14,15);
    v16bf bv0 = *(const v16bf*)(bp0 + kt);
    v16bf bv1 = *(const v16bf*)(bp1 + kt);
    acc0 = __builtin_amdgcn_wmma_f32_16x16x32_bf16(false, av, false, bv0,
                                                   (short)0, acc0, false, false);
    acc1 = __builtin_amdgcn_wmma_f32_16x16x32_bf16(false, av, false, bv1,
                                                   (short)0, acc1, false, false);
  }
  store_tile(C, bias, residual, acc0, mt, cb0, lane, ldc, act);
  store_tile(C, bias, residual, acc1, mt, cb1, lane, ldc, act);
}

// ---------------------------------------------------------------------------
// Generic GEMM (fp32 A, guarded tails). Used only for W_red (M=B*L,N=30,K=80).
// ---------------------------------------------------------------------------
__global__ void __launch_bounds__(256)
gemm_bf16_gen(const float* __restrict__ A, const unsigned short* __restrict__ Bt,
              const float* __restrict__ bias, float* __restrict__ C,
              int M, int N, int K, int ldc)
{
  int wave = (int)((blockIdx.x * blockDim.x + threadIdx.x) >> 5);
  int lane = threadIdx.x & 31;
  int ntN = (N + 15) >> 4;
  int ntM = (M + 15) >> 4;
  if (wave >= ntM * ntN) return;
  int mt = wave / ntN, nt = wave % ntN;

  int ra    = mt * 16 + (lane & 15);
  int abase = (lane >> 4) * 16;
  int cb    = nt * 16 + (lane & 15);
  int bbase = (lane >> 4) * 16;

  v8f acc = {0.f,0.f,0.f,0.f,0.f,0.f,0.f,0.f};
  for (int kt = 0; kt < K; kt += 32) {
    v16bf av, bv;
#pragma unroll
    for (int e = 0; e < 16; ++e) {
      int k = kt + abase + e;
      float v = (ra < M && k < K) ? A[(size_t)ra * K + k] : 0.f;
      av[e] = f2bf(v);
    }
#pragma unroll
    for (int e = 0; e < 16; ++e) {
      int k = kt + bbase + e;
      bv[e] = (cb < N && k < K) ? us2bf(Bt[(size_t)cb * K + k]) : us2bf(0);
    }
    acc = __builtin_amdgcn_wmma_f32_16x16x32_bf16(false, av, false, bv,
                                                  (short)0, acc, false, false);
  }
  if (cb < N) {
    float bvv = bias ? bias[cb] : 0.f;
#pragma unroll
    for (int v = 0; v < 8; ++v) {
      int cr = mt * 16 + v + ((lane >> 4) << 3);
      if (cr < M) C[(size_t)cr * ldc + cb] = acc[v] + bvv;
    }
  }
}

// ---- fp32 [K][N] -> bf16 [N][K] (transpose) -------------------------------
__global__ void __launch_bounds__(256)
convT_k(const float* __restrict__ W, unsigned short* __restrict__ O, int K, int N)
{
  int i = blockIdx.x * 256 + threadIdx.x;
  if (i >= K * N) return;
  int k = i / N, n = i % N;
  O[(size_t)n * K + k] = f2us(W[i]);
}
// ---- fp32 -> bf16, same layout --------------------------------------------
__global__ void __launch_bounds__(256)
conv_k(const float* __restrict__ W, unsigned short* __restrict__ O, int n)
{
  int i = blockIdx.x * 256 + threadIdx.x;
  if (i < n) O[i] = f2us(W[i]);
}

__global__ void __launch_bounds__(256)
fill_k(float* __restrict__ p, float v, int n)
{
  int i = blockIdx.x * 256 + threadIdx.x;
  if (i < n) p[i] = v;
}

// ---- GCN pieces -----------------------------------------------------------
__global__ void __launch_bounds__(256)
deg_edges_k(const int* __restrict__ dst, float* __restrict__ deg, int E)
{
  int e = blockIdx.x * 256 + threadIdx.x;
  if (e < E) atomicAdd(&deg[dst[e]], 1.f);
}
__global__ void __launch_bounds__(256)
rsqrt_k(float* __restrict__ p, int n)
{
  int i = blockIdx.x * 256 + threadIdx.x;
  if (i < n) p[i] = rsqrtf(p[i]);
}
__global__ void __launch_bounds__(256)
accinit_k(const float* __restrict__ xw, const float* __restrict__ dinv,
          const float* __restrict__ b, float* __restrict__ acc, int Nn, int F)
{
  unsigned i = blockIdx.x * 256u + threadIdx.x;
  if (i >= (unsigned)Nn * (unsigned)F) return;
  int n = i / F, f = i % F;
  float di = dinv[n];
  acc[i] = xw[i] * di * di + b[f];
}
__global__ void __launch_bounds__(256)
scatter_k(const float* __restrict__ xw, const int* __restrict__ src,
          const int* __restrict__ dst, const float* __restrict__ dinv,
          float* __restrict__ acc, int E, int F)
{
  unsigned i = blockIdx.x * 256u + threadIdx.x;
  if (i >= (unsigned)E * (unsigned)F) return;
  int e = i / F, f = i % F;
  int s = src[e], d = dst[e];
  float nrm = dinv[s] * dinv[d];
  atomicAdd(&acc[(size_t)d * F + f], xw[(size_t)s * F + f] * nrm);
}
__global__ void __launch_bounds__(256)
count_k(const int* __restrict__ batch, float* __restrict__ cnt, int Nn)
{
  int i = blockIdx.x * 256 + threadIdx.x;
  if (i < Nn) atomicAdd(&cnt[batch[i]], 1.f);
}
__global__ void __launch_bounds__(256)
poolsum_k(const float* __restrict__ acc, const int* __restrict__ batch,
          float* __restrict__ pooled, int Nn, int F)
{
  unsigned i = blockIdx.x * 256u + threadIdx.x;
  if (i >= (unsigned)Nn * (unsigned)F) return;
  int n = i / F, f = i % F;
  float v = acc[i];
  v = (v >= 0.f) ? v : NEG_SLOPE * v;
  atomicAdd(&pooled[(size_t)batch[n] * F + f], v);
}
__global__ void __launch_bounds__(256)
pooldiv_k(float* __restrict__ pooled, const float* __restrict__ cnt, int Bb, int F)
{
  int i = blockIdx.x * 256 + threadIdx.x;
  if (i >= Bb * F) return;
  int b = i / F;
  pooled[i] /= fmaxf(cnt[b], 1.f);
}

// ---- descriptor assemble: redtmp[4][B*L][32] (cols 0..29 valid) -> mas[S][B][32]
__global__ void __launch_bounds__(256)
assemble_k(const float* __restrict__ redtmp, float* __restrict__ mas, int Bb, int L)
{
  int total = 4 * Bb * L * 32;
  int i = blockIdx.x * 256 + threadIdx.x;
  if (i >= total) return;
  int per = Bb * L * 32;
  int c = i / per;
  int rem = i % per;
  int r = rem / 32, d = rem % 32;
  int b = r / L, l = r % L;
  int s = c * L + l;
  float v;
  if (d < 30)       v = redtmp[i];
  else if (d == 30) v = (c == 0 || c == 2) ? 1.f : 0.f;  // straight flag
  else              v = (c < 2) ? 1.f : 0.f;             // first flag
  mas[((size_t)s * Bb + b) * 32 + d] = v;
}

// ---- attention: one block per (b,h); K/V staged in LDS --------------------
__global__ void __launch_bounds__(256)
attention_k(const float* __restrict__ qkv /*[S][B][96]*/,
            float* __restrict__ outp /*[S][B][32]*/, int S, int Bb)
{
  __shared__ float ksh[512 * 8];
  __shared__ float vsh[512 * 8];
  int b = blockIdx.x >> 2;
  int h = blockIdx.x & 3;
  int tid = threadIdx.x;
  for (int i = tid; i < S * 8; i += 256) {
    int s = i >> 3, d = i & 7;
    size_t base = ((size_t)s * Bb + b) * 96 + h * 8 + d;
    ksh[i] = qkv[base + 32];
    vsh[i] = qkv[base + 64];
  }
  __syncthreads();
  const float scale = 0.3535533906f;  // 8^-0.5
  for (int sq = tid; sq < S; sq += 256) {
    float q[8];
    size_t qb = ((size_t)sq * Bb + b) * 96 + h * 8;
#pragma unroll
    for (int d = 0; d < 8; ++d) q[d] = qkv[qb + d];
    float mx = -3.4e38f;
    for (int t = 0; t < S; ++t) {
      float dot = 0.f;
#pragma unroll
      for (int d = 0; d < 8; ++d) dot += q[d] * ksh[t * 8 + d];
      mx = fmaxf(mx, dot * scale);
    }
    float sum = 0.f, o[8] = {0.f,0.f,0.f,0.f,0.f,0.f,0.f,0.f};
    for (int t = 0; t < S; ++t) {
      float dot = 0.f;
#pragma unroll
      for (int d = 0; d < 8; ++d) dot += q[d] * ksh[t * 8 + d];
      float w = __expf(dot * scale - mx);
      sum += w;
#pragma unroll
      for (int d = 0; d < 8; ++d) o[d] += w * vsh[t * 8 + d];
    }
    float inv = 1.f / sum;
    size_t ob = ((size_t)sq * Bb + b) * 32 + h * 8;
#pragma unroll
    for (int d = 0; d < 8; ++d) outp[ob + d] = o[d] * inv;
  }
}

// ---- LayerNorm over D=32: one wave32 per row ------------------------------
__global__ void __launch_bounds__(256)
ln_k(const float* __restrict__ in, const float* __restrict__ g,
     const float* __restrict__ bta, float* __restrict__ outp, int rows)
{
  int wid = (int)((blockIdx.x * 256 + threadIdx.x) >> 5);
  int lane = threadIdx.x & 31;
  if (wid >= rows) return;
  float x = in[(size_t)wid * 32 + lane];
  float s = x;
  for (int m = 16; m >= 1; m >>= 1) s += __shfl_xor(s, m, 32);
  float mean = s * (1.f / 32.f);
  float d = x - mean;
  float v = d * d;
  for (int m = 16; m >= 1; m >>= 1) v += __shfl_xor(v, m, 32);
  v *= (1.f / 32.f);
  outp[(size_t)wid * 32 + lane] = d * rsqrtf(v + LN_EPS) * g[lane] + bta[lane];
}

__global__ void __launch_bounds__(256)
masmean_k(const float* __restrict__ mas, float* __restrict__ outp, int S, int Bb)
{
  int i = blockIdx.x * 256 + threadIdx.x;
  if (i >= Bb * 32) return;
  int b = i / 32, d = i % 32;
  float sum = 0.f;
  for (int s = 0; s < S; ++s) sum += mas[((size_t)s * Bb + b) * 32 + d];
  outp[i] = sum / (float)S;
}

__global__ void
final_k(const float* __restrict__ xa, const float* __restrict__ xb,
        const float* __restrict__ mm, const float* __restrict__ Wf,
        const float* __restrict__ bf, float* __restrict__ outp, int Bb)
{
  int b = threadIdx.x;
  if (b >= Bb) return;
  float s = bf[0];
  for (int j = 0; j < 128; ++j) s += xa[b * 128 + j] * Wf[j];
  for (int j = 0; j < 128; ++j) s += xb[b * 128 + j] * Wf[128 + j];
  for (int j = 0; j < 32; ++j)  s += mm[b * 32 + j] * Wf[256 + j];
  outp[b] = s;
}

// ---------------------------------------------------------------------------
extern "C" void kernel_launch(void* const* d_in, const int* in_sizes, int n_in,
                              void* d_out, int out_size, void* d_ws, size_t ws_size,
                              hipStream_t stream)
{
  const int F = 1024, L = 128, Dsc = 80, Dm = 32, FFd = 128, OUT = 128, H = 4;
  const int Nn = in_sizes[0] / F;
  const int E  = in_sizes[1] / 2;
  const int Bb = in_sizes[6] / (L * Dsc);
  const int S  = 4 * L;
  const int SB = S * Bb;
  const int NL = in_sizes[20] / (3 * Dm * Dm);

  const float* x1   = (const float*)d_in[0];
  const int*   ei1  = (const int*)d_in[1];
  const int*   bat1 = (const int*)d_in[2];
  const float* x2   = (const float*)d_in[3];
  const int*   ei2  = (const int*)d_in[4];
  const int*   bat2 = (const int*)d_in[5];
  const float* masi[4] = {(const float*)d_in[6], (const float*)d_in[7],
                          (const float*)d_in[8], (const float*)d_in[9]};
  const float* Wc1 = (const float*)d_in[10]; const float* bc1 = (const float*)d_in[11];
  const float* Wf1 = (const float*)d_in[12]; const float* bfc1 = (const float*)d_in[13];
  const float* Wc2 = (const float*)d_in[14]; const float* bc2 = (const float*)d_in[15];
  const float* Wf2 = (const float*)d_in[16]; const float* bfc2 = (const float*)d_in[17];
  const float* Wred = (const float*)d_in[18]; const float* bred = (const float*)d_in[19];
  const float* Wqkv = (const float*)d_in[20]; const float* bqkv = (const float*)d_in[21];
  const float* Wo   = (const float*)d_in[22]; const float* bo   = (const float*)d_in[23];
  const float* ln1g = (const float*)d_in[24]; const float* ln1b = (const float*)d_in[25];
  const float* ln2g = (const float*)d_in[26]; const float* ln2b = (const float*)d_in[27];
  const float* Wff1 = (const float*)d_in[28]; const float* bff1 = (const float*)d_in[29];
  const float* Wff2 = (const float*)d_in[30]; const float* bff2 = (const float*)d_in[31];
  const float* Wfin = (const float*)d_in[32]; const float* bfin = (const float*)d_in[33];

  // ---- workspace carve ----
  char* w = (char*)d_ws; size_t off = 0;
  auto carve = [&](size_t bytes) -> void* {
    void* p = w + off; off = (off + bytes + 255) & ~(size_t)255; return p;
  };
  unsigned short* WT1   = (unsigned short*)carve((size_t)F * F * 2);
  unsigned short* WT2   = (unsigned short*)carve((size_t)F * F * 2);
  unsigned short* WTfc1 = (unsigned short*)carve((size_t)OUT * F * 2);
  unsigned short* WTfc2 = (unsigned short*)carve((size_t)OUT * F * 2);
  unsigned short* WTred = (unsigned short*)carve((size_t)30 * Dsc * 2);
  unsigned short* WTqkv = (unsigned short*)carve((size_t)NL * 3 * Dm * Dm * 2);
  unsigned short* WTo   = (unsigned short*)carve((size_t)NL * Dm * Dm * 2);
  unsigned short* WTff1 = (unsigned short*)carve((size_t)NL * FFd * Dm * 2);
  unsigned short* WTff2 = (unsigned short*)carve((size_t)NL * Dm * FFd * 2);
  unsigned short* Abig  = (unsigned short*)carve((size_t)Nn * F * 2);   // bf16 A (GCN x)
  unsigned short* Asml  = (unsigned short*)carve((size_t)SB * FFd * 2); // bf16 A (small GEMMs)
  float* deg    = (float*)carve((size_t)Nn * 4);
  float* cnt    = (float*)carve((size_t)Bb * 4);
  float* pooled = (float*)carve((size_t)Bb * F * 4);
  float* xfc1   = (float*)carve((size_t)Bb * OUT * 4);
  float* xfc2   = (float*)carve((size_t)Bb * OUT * 4);
  float* mmean  = (float*)carve((size_t)Bb * Dm * 4);
  float* mas    = (float*)carve((size_t)SB * Dm * 4);
  float* tmpA   = (float*)carve((size_t)SB * FFd * 4);   // qkv / ff hidden
  float* tmpB   = (float*)carve((size_t)SB * Dm * 4);    // redtmp / attn out
  float* tmpC   = (float*)carve((size_t)SB * Dm * 4);    // pre-LN residual sum
  float* xw     = (float*)carve((size_t)Nn * F * 4);
  float* acc    = (float*)carve((size_t)Nn * F * 4);
  (void)ws_size; (void)n_in; (void)out_size;

  auto launch_fast = [&](const unsigned short* Ab, const unsigned short* Bt,
                         const float* bias, const float* residual, float* C,
                         int M, int N, int K, int ldc, int act) {
    int tiles = (M >> 4) * (N >> 5);
    gemm_bf16_fast<<<ceil_div_h(tiles, 8), 256, 0, stream>>>(
        Ab, Bt, bias, residual, C, M, N, K, ldc, act);
  };
  auto convA = [&](const float* src, unsigned short* dst, long long n) {
    conv_k<<<ceil_div_h(n, 256), 256, 0, stream>>>(src, dst, (int)n);
  };

  // ---- weight conversions (once per launch; cheap) ----
  convT_k<<<ceil_div_h((long long)F * F, 256), 256, 0, stream>>>(Wc1, WT1, F, F);
  convT_k<<<ceil_div_h((long long)F * F, 256), 256, 0, stream>>>(Wc2, WT2, F, F);
  convT_k<<<ceil_div_h((long long)F * OUT, 256), 256, 0, stream>>>(Wf1, WTfc1, F, OUT);
  convT_k<<<ceil_div_h((long long)F * OUT, 256), 256, 0, stream>>>(Wf2, WTfc2, F, OUT);
  convT_k<<<ceil_div_h((long long)Dsc * 30, 256), 256, 0, stream>>>(Wred, WTred, Dsc, 30);
  convA(Wqkv, WTqkv, (long long)NL * 3 * Dm * Dm);
  convA(Wo,   WTo,   (long long)NL * Dm * Dm);
  convA(Wff1, WTff1, (long long)NL * FFd * Dm);
  convA(Wff2, WTff2, (long long)NL * Dm * FFd);

  // ---- GCN branch (run twice, reusing Abig/xw/acc/deg/pooled) ----
  auto run_branch = [&](const float* X, const int* ei, const int* bat,
                        const unsigned short* WTc, const float* bconv,
                        const unsigned short* WTf, const float* bfc, float* xfc) {
    convA(X, Abig, (long long)Nn * F);
    launch_fast(Abig, WTc, nullptr, nullptr, xw, Nn, F, F, F, 0);
    fill_k<<<ceil_div_h(Nn, 256), 256, 0, stream>>>(deg, 1.f, Nn);
    deg_edges_k<<<ceil_div_h(E, 256), 256, 0, stream>>>(ei + E, deg, E);
    rsqrt_k<<<ceil_div_h(Nn, 256), 256, 0, stream>>>(deg, Nn);
    accinit_k<<<ceil_div_h((long long)Nn * F, 256), 256, 0, stream>>>(xw, deg, bconv, acc, Nn, F);
    scatter_k<<<ceil_div_h((long long)E * F, 256), 256, 0, stream>>>(xw, ei, ei + E, deg, acc, E, F);
    fill_k<<<ceil_div_h((long long)Bb * F, 256), 256, 0, stream>>>(pooled, 0.f, Bb * F);
    fill_k<<<1, 256, 0, stream>>>(cnt, 0.f, Bb);
    count_k<<<ceil_div_h(Nn, 256), 256, 0, stream>>>(bat, cnt, Nn);
    poolsum_k<<<ceil_div_h((long long)Nn * F, 256), 256, 0, stream>>>(acc, bat, pooled, Nn, F);
    pooldiv_k<<<ceil_div_h((long long)Bb * F, 256), 256, 0, stream>>>(pooled, cnt, Bb, F);
    convA(pooled, Asml, (long long)Bb * F);
    launch_fast(Asml, WTf, bfc, nullptr, xfc, Bb, OUT, F, OUT, 1);
  };
  run_branch(x1, ei1, bat1, WT1, bc1, WTfc1, bfc1, xfc1);
  run_branch(x2, ei2, bat2, WT2, bc2, WTfc2, bfc2, xfc2);

  // ---- descriptor reduce (generic kernel: K=80, N=30) + tag assemble ----
  for (int c = 0; c < 4; ++c) {
    int tiles = ceil_div_h(Bb * L, 16) * ceil_div_h(30, 16);
    gemm_bf16_gen<<<ceil_div_h(tiles, 8), 256, 0, stream>>>(
        masi[c], WTred, bred, tmpB + (size_t)c * Bb * L * 32, Bb * L, 30, Dsc, 32);
  }
  assemble_k<<<ceil_div_h((long long)4 * Bb * L * 32, 256), 256, 0, stream>>>(tmpB, mas, Bb, L);

  // ---- transformer encoder layers ----
  for (int l = 0; l < NL; ++l) {
    convA(mas, Asml, (long long)SB * Dm);
    launch_fast(Asml, WTqkv + (size_t)l * 3 * Dm * Dm, bqkv + l * 3 * Dm, nullptr,
                tmpA, SB, 3 * Dm, Dm, 3 * Dm, 0);
    attention_k<<<Bb * H, 256, 0, stream>>>(tmpA, tmpB, S, Bb);
    convA(tmpB, Asml, (long long)SB * Dm);
    launch_fast(Asml, WTo + (size_t)l * Dm * Dm, bo + l * Dm, mas,
                tmpC, SB, Dm, Dm, Dm, 0);
    ln_k<<<ceil_div_h(SB, 8), 256, 0, stream>>>(tmpC, ln1g + l * Dm, ln1b + l * Dm, mas, SB);
    convA(mas, Asml, (long long)SB * Dm);
    launch_fast(Asml, WTff1 + (size_t)l * FFd * Dm, bff1 + l * FFd, nullptr,
                tmpA, SB, FFd, Dm, FFd, 2);
    convA(tmpA, Asml, (long long)SB * FFd);
    launch_fast(Asml, WTff2 + (size_t)l * Dm * FFd, bff2 + l * Dm, mas,
                tmpC, SB, Dm, FFd, Dm, 0);
    ln_k<<<ceil_div_h(SB, 8), 256, 0, stream>>>(tmpC, ln2g + l * Dm, ln2b + l * Dm, mas, SB);
  }

  masmean_k<<<ceil_div_h((long long)Bb * Dm, 256), 256, 0, stream>>>(mas, mmean, S, Bb);
  final_k<<<1, Bb, 0, stream>>>(xfc1, xfc2, mmean, Wfin, bfin, (float*)d_out, Bb);
}
